// ModelNetPointNet2_34308198761135
// MI455X (gfx1250) — matrix operand order repaired and checked by reading, not compiled, not run on hardware
//
#include <hip/hip_runtime.h>
#include <cstdint>
#include <cstddef>

typedef __attribute__((ext_vector_type(16))) _Float16 v16h;
typedef __attribute__((ext_vector_type(8)))  _Float16 v8h;
typedef __attribute__((ext_vector_type(8)))  float    v8f;
typedef __attribute__((ext_vector_type(4)))  unsigned int u32x4;
typedef __attribute__((ext_vector_type(8)))  int      i32x8;
typedef __attribute__((ext_vector_type(4)))  int      i32x4;

// ---------------------------------------------------------------------------
// float atomic max via int/uint atomics (valid for all floats, init -FLT_MAX)
// ---------------------------------------------------------------------------
__device__ inline void atomicMaxFloat(float* addr, float v) {
  if (v >= 0.0f) atomicMax((int*)addr, __float_as_int(v));
  else           atomicMin((unsigned int*)addr, __float_as_uint(v));
}

// ---------------------------------------------------------------------------
// A fragment (16x32 f16) from an LDS panel, ISA 7.12.2 layout, vectorized:
// lane<16: row=lane, a[0..7]=K(kt+0..7),  a[8..15]=K(kt+16..23)
// lane>=16: row=lane-16, a[0..7]=K(kt+8..15), a[8..15]=K(kt+24..31)
// ---------------------------------------------------------------------------
__device__ inline v16h load_a_frag(const _Float16* __restrict__ p, int ld, int kt, int lane) {
  int m  = lane & 15;
  int kb = (lane >> 4) * 8;
  const _Float16* r = p + m * ld + kt + kb;
  v8h lo = *(const v8h*)(r);        // 16B ds/vmem load
  v8h hi = *(const v8h*)(r + 16);   // 16B ds/vmem load
  return __builtin_shufflevector(lo, hi, 0, 1, 2, 3, 4, 5, 6, 7, 8, 9, 10, 11, 12, 13, 14, 15);
}

// B fragment from packed weights: fragment-major layout, lane's 16 halfs are
// contiguous (32B). idx = ((kt32*nTiles + nt)*32 + lane)*16
__device__ inline v16h load_b_packed(const _Float16* __restrict__ Wp, int nTiles, int kt32, int nt, int lane) {
  return *(const v16h*)(Wp + (((size_t)kt32 * nTiles + nt) * 32 + lane) * 16);
}

// ---------------------------------------------------------------------------
// One MLP layer on a 16-row panel: out = relu(in @ W + bias), f32 accumulate.
// inP: [16][CinPad] f16 in LDS. Wp: packed fragments (LDS or global).
// ---------------------------------------------------------------------------
__device__ inline void mlp_layer(const _Float16* __restrict__ Wp, const float* __restrict__ bias,
                                 const _Float16* inP, _Float16* outP,
                                 int CinPad, int Cout, int lane) {
  int col = lane & 15, rbase = (lane >> 4) * 8;
  int nT = Cout >> 4, kT = CinPad >> 5;
  for (int nt = 0; nt < nT; ++nt) {
    v8f acc = {};
    for (int kt = 0; kt < kT; ++kt) {
      v16h a = load_a_frag(inP, CinPad, kt * 32, lane);
      v16h b = load_b_packed(Wp, nT, kt, nt, lane);
      acc = __builtin_amdgcn_wmma_f32_16x16x32_f16(false, a, false, b, (short)0, acc, false, false);
    }
    float bv = bias[nt * 16 + col];
#pragma unroll
    for (int r = 0; r < 8; ++r) {
      float v = fmaxf(acc[r] + bv, 0.0f);
      outP[(rbase + r) * Cout + nt * 16 + col] = (_Float16)v;
    }
  }
}

// Final (plain) layer: masked atomic-max into outRow[Cout].
__device__ inline void mlp_final_max(const _Float16* __restrict__ Wp, const float* __restrict__ bias,
                                     const _Float16* inP, int CinPad, int Cout,
                                     unsigned rowmask, float* __restrict__ outRow, int lane) {
  int col = lane & 15, rbase = (lane >> 4) * 8;
  int nT = Cout >> 4, kT = CinPad >> 5;
  for (int nt = 0; nt < nT; ++nt) {
    v8f acc = {};
    for (int kt = 0; kt < kT; ++kt) {
      v16h a = load_a_frag(inP, CinPad, kt * 32, lane);
      v16h b = load_b_packed(Wp, nT, kt, nt, lane);
      acc = __builtin_amdgcn_wmma_f32_16x16x32_f16(false, a, false, b, (short)0, acc, false, false);
    }
    float bv = bias[nt * 16 + col];
#pragma unroll
    for (int r = 0; r < 8; ++r) {
      if ((rowmask >> (rbase + r)) & 1u)
        atomicMaxFloat(outRow + nt * 16 + col, acc[r] + bv);
    }
  }
}

// ---------------------------------------------------------------------------
// TDM: stage a packed f16 weight array (nhalf % 4 == 0) into LDS as a 1-D
// tensor of 8-byte elements. D# per ISA 8.3/8.4: count=1, type=2 ("image"),
// data_size=3 (8B), tensor_dim0 = tile_dim0 = stride0 = nhalf/4.
// This toolchain's builtin is the 6-arg form (g0, g1, g2, g3, g4, cpol).
// ---------------------------------------------------------------------------
__device__ inline void tdm_load_weights(const _Float16* src, unsigned lds_off_bytes, int nhalf) {
#if defined(__gfx1250__) && __has_builtin(__builtin_amdgcn_tensor_load_to_lds)
  unsigned n8 = (unsigned)nhalf >> 2;                       // 8-byte elements
  unsigned long long ga = (unsigned long long)(const void*)src;
  u32x4 g0 = { 1u,                                          // count=1 (valid user D#)
               lds_off_bytes,                               // lds_addr
               (unsigned)(ga & 0xFFFFFFFFu),                // global_addr[31:0]
               (unsigned)((ga >> 32) & 0x01FFFFFFu) | 0x80000000u }; // addr[56:32] | type=2
  i32x8 g1;
  g1[0] = (int)(3u << 16);                                  // workgroup_mask=0, data_size=8B
  g1[1] = (int)((n8 & 0xFFFFu) << 16);                      // tensor_dim0[15:0] in bits 63:48
  g1[2] = (int)(((n8 >> 16) & 0xFFFFu) | (1u << 16));       // tensor_dim0[31:16], tensor_dim1=1
  g1[3] = (int)((n8 & 0xFFFFu) << 16);                      // tile_dim0 in bits 127:112
  g1[4] = 0;                                                // tile_dim1/2 unused
  g1[5] = (int)n8;                                          // tensor_dim0_stride[31:0]
  g1[6] = 0;
  g1[7] = 0;
  i32x4 z4 = { 0, 0, 0, 0 };
  i32x8 z8 = { 0, 0, 0, 0, 0, 0, 0, 0 };
  __builtin_amdgcn_tensor_load_to_lds(g0, g1, z4, z4, z8, 0);
#else
  (void)src; (void)lds_off_bytes; (void)nhalf;
#endif
}

// Stage three packed weight arrays into LDS (cooperative, then barrier).
__device__ inline void stage_weights(_Float16* lw, const _Float16* W1, int s1,
                                     const _Float16* W2, int s2,
                                     const _Float16* W3, int s3,
                                     int wib, int nthr) {
#if defined(__gfx1250__) && __has_builtin(__builtin_amdgcn_tensor_load_to_lds) && \
    __has_builtin(__builtin_amdgcn_s_wait_tensorcnt)
  if (wib == 0) {                       // one wave issues the three TDM descriptors
    unsigned base = (unsigned)__builtin_amdgcn_groupstaticsize();
    tdm_load_weights(W1, base, s1);
    tdm_load_weights(W2, base + (unsigned)s1 * 2u, s2);
    tdm_load_weights(W3, base + (unsigned)(s1 + s2) * 2u, s3);
    __builtin_amdgcn_s_wait_tensorcnt(0);
  }
  (void)lw; (void)nthr;
#else
  int tid = threadIdx.x;
  for (int i = tid * 8; i < s1; i += nthr * 8) *(v8h*)(lw + i)           = *(const v8h*)(W1 + i);
  for (int i = tid * 8; i < s2; i += nthr * 8) *(v8h*)(lw + s1 + i)      = *(const v8h*)(W2 + i);
  for (int i = tid * 8; i < s3; i += nthr * 8) *(v8h*)(lw + s1 + s2 + i) = *(const v8h*)(W3 + i);
#endif
  __syncthreads();
}

// ---------------------------------------------------------------------------
// Weight prep: f32 [Cin][Cout] -> packed f16 fragments [kt32][nt][lane][16],
// zero padded to [CinPad][CoutPad]. One thread produces one 16-half fragment.
// ---------------------------------------------------------------------------
__global__ void convert_w_packed(const float* __restrict__ W, _Float16* __restrict__ out,
                                 int Cin, int Cout, int CinPad, int CoutPad) {
  int i = blockIdx.x * blockDim.x + threadIdx.x;
  int nT = CoutPad >> 4;
  int nfrag = (CinPad >> 5) * nT * 32;
  if (i >= nfrag) return;
  int lane = i & 31;
  int t = i >> 5;
  int nt = t % nT, kt32 = t / nT;
  int col = lane & 15, ko = (lane >> 4) * 8;
  int n = nt * 16 + col;
  v16h vals;
#pragma unroll
  for (int e = 0; e < 16; ++e) {
    int k = kt32 * 32 + ko + (e & 7) + (e >> 3) * 16;
    float v = (k < Cin && n < Cout) ? W[(size_t)k * Cout + n] : 0.0f;
    vals[e] = (_Float16)v;
  }
  *(v16h*)(out + (size_t)i * 16) = vals;
}

__global__ void fill_kernel(float* __restrict__ p, float v, int n) {
  int i = blockIdx.x * blockDim.x + threadIdx.x;
  if (i < n) p[i] = v;
}

// ---------------------------------------------------------------------------
// Farthest point sampling: one workgroup (256 threads) per batch item.
// ---------------------------------------------------------------------------
__global__ void fps_kernel(const float* __restrict__ pos, int P, int S, int* __restrict__ idx) {
  extern __shared__ char smem[];
  float* d  = (float*)smem;
  float* rv = d + P;
  int*   ri = (int*)(rv + 256);
  __shared__ float bp[3];
  int b = blockIdx.x, tid = threadIdx.x;
  const float* pb = pos + (size_t)b * P * 3;
  float x0 = pb[0], y0 = pb[1], z0 = pb[2];
  for (int p = tid; p < P; p += 256) {
    float dx = pb[p * 3] - x0, dy = pb[p * 3 + 1] - y0, dz = pb[p * 3 + 2] - z0;
    d[p] = dx * dx + dy * dy + dz * dz;
  }
  if (tid == 0) idx[(size_t)b * S] = 0;
  __syncthreads();
  for (int i = 1; i < S; ++i) {
    float bv = -1.0f; int bi = 0;
    for (int p = tid; p < P; p += 256)
      if (d[p] > bv) { bv = d[p]; bi = p; }
    rv[tid] = bv; ri[tid] = bi;
    __syncthreads();
    for (int off = 128; off > 0; off >>= 1) {
      if (tid < off && rv[tid + off] > rv[tid]) { rv[tid] = rv[tid + off]; ri[tid] = ri[tid + off]; }
      __syncthreads();
    }
    int sel = ri[0];
    if (tid == 0) {
      idx[(size_t)b * S + i] = sel;
      bp[0] = pb[sel * 3]; bp[1] = pb[sel * 3 + 1]; bp[2] = pb[sel * 3 + 2];
    }
    __syncthreads();
    float sx = bp[0], sy = bp[1], sz = bp[2];
    for (int p = tid; p < P; p += 256) {
      float dx = pb[p * 3] - sx, dy = pb[p * 3 + 1] - sy, dz = pb[p * 3 + 2] - sz;
      float nd = dx * dx + dy * dy + dz * dz;
      if (nd < d[p]) d[p] = nd;
    }
    __syncthreads();
  }
}

__global__ void gather_pos_kernel(const float* __restrict__ pos, const int* __restrict__ idx,
                                  int P, int S, int NQ, float* __restrict__ out) {
  int i = blockIdx.x * blockDim.x + threadIdx.x;
  if (i >= NQ) return;
  int b = i / S;
  int j = idx[i];
  const float* p = pos + ((size_t)b * P + j) * 3;
  out[i * 3 + 0] = p[0];
  out[i * 3 + 1] = p[1];
  out[i * 3 + 2] = p[2];
}

// ---------------------------------------------------------------------------
// 64-NN: one wave32 per query, iterative argmin over an LDS distance array.
// block = 128 threads = 4 waves. dyn smem: 4 * P floats.
// ---------------------------------------------------------------------------
__global__ void knn_kernel(const float* __restrict__ pos, int P,
                           const float* __restrict__ qpos, int S, int NQ,
                           int* __restrict__ nb, float* __restrict__ d2k) {
  extern __shared__ char smem[];
  int lane = threadIdx.x & 31;
  int wib  = threadIdx.x >> 5;
  int q = blockIdx.x * 4 + wib;
  if (q >= NQ) return;
  float* d = (float*)smem + (size_t)wib * P;
  int b = q / S;
  const float* pb = pos + (size_t)b * P * 3;
  float qx = qpos[q * 3], qy = qpos[q * 3 + 1], qz = qpos[q * 3 + 2];
  for (int p = lane; p < P; p += 32) {
    float dx = pb[p * 3] - qx, dy = pb[p * 3 + 1] - qy, dz = pb[p * 3 + 2] - qz;
    d[p] = dx * dx + dy * dy + dz * dz;
  }
  __builtin_amdgcn_wave_barrier();
  for (int j = 0; j < 64; ++j) {
    float bv = 3.4e38f; int bi = 0;
    for (int p = lane; p < P; p += 32) {
      float v = d[p];
      if (v < bv) { bv = v; bi = p; }
    }
#pragma unroll
    for (int off = 16; off > 0; off >>= 1) {
      float ov = __shfl_down(bv, off);
      int   oi = __shfl_down(bi, off);
      if (ov < bv) { bv = ov; bi = oi; }
    }
    bi = __shfl(bi, 0);
    bv = __shfl(bv, 0);
    if (lane == 0) {
      nb[(size_t)q * 64 + j]  = bi;
      d2k[(size_t)q * 64 + j] = bv;
      d[bi] = 3.4e38f;
    }
    __builtin_amdgcn_wave_barrier();
  }
}

// ---------------------------------------------------------------------------
// Fused set-abstraction MLP. One wave per 16-neighbor tile (K=64 -> 4 tiles).
// Weights TDM-staged into LDS once per block; 3 WMMA layers fully in LDS;
// masked float-atomic-max into out[q][C3].
// LDS layout: [W1|W2|W3 packed f16][4 waves * 2 panels * 16*PanelC f16]
// ---------------------------------------------------------------------------
__global__ void sa_mlp_kernel(const float* __restrict__ pos, int P,
                              const float* __restrict__ xin, int Cx,
                              const int* __restrict__ sampidx, int S,
                              const int* __restrict__ nb, const float* __restrict__ d2k, float r2,
                              const _Float16* __restrict__ W1, const float* __restrict__ b1, int C1, int W1sz,
                              const _Float16* __restrict__ W2, const float* __restrict__ b2, int C2, int W2sz,
                              const _Float16* __restrict__ W3, const float* __restrict__ b3, int C3, int W3sz,
                              int CinPad, int PanelC,
                              float* __restrict__ out) {
  extern __shared__ char smem[];
  int lane = threadIdx.x & 31;
  int wib  = threadIdx.x >> 5;
  _Float16* lw = (_Float16*)smem;
  _Float16* lw1 = lw;
  _Float16* lw2 = lw1 + W1sz;
  _Float16* lw3 = lw2 + W2sz;
  _Float16* inP  = lw3 + W3sz + (size_t)wib * 2 * 16 * PanelC;
  _Float16* outP = inP + 16 * PanelC;

  stage_weights(lw, W1, W1sz, W2, W2sz, W3, W3sz, wib, 128);

  int w  = blockIdx.x * 4 + wib;   // global tile id
  int q  = w >> 2;                 // query id in [0, B*S)
  int tk = w & 3;                  // which 16 of K=64 neighbors
  int b  = q / S;
  int si = sampidx[q];
  const float* pb = pos + (size_t)b * P * 3;
  float qx = pb[si * 3], qy = pb[si * 3 + 1], qz = pb[si * 3 + 2];
  const int*   nbq = nb + (size_t)q * 64 + tk * 16;
  const float* dq  = d2k + (size_t)q * 64 + tk * 16;

  int Cfeat = Cx + 3;
  for (int e = lane; e < 16 * CinPad; e += 32) {
    int r = e / CinPad, c = e - r * CinPad;
    float v = 0.0f;
    if (c < Cfeat) {
      int j = nbq[r];
      if (c < Cx) {
        v = xin[((size_t)b * P + j) * Cx + c];
      } else {
        int dd = c - Cx;
        v = pb[j * 3 + dd] - (dd == 0 ? qx : (dd == 1 ? qy : qz));
      }
    }
    inP[r * CinPad + c] = (_Float16)v;
  }
  unsigned rowmask = 0;
  for (int r = 0; r < 16; ++r)
    if (dq[r] <= r2) rowmask |= (1u << r);
  __builtin_amdgcn_wave_barrier();

  mlp_layer(lw1, b1, inP, outP, CinPad, C1, lane);
  mlp_layer(lw2, b2, outP, inP, C1, C2, lane);
  mlp_final_max(lw3, b3, inP, C2, C3, rowmask, out + (size_t)q * C3, lane);
}

// ---------------------------------------------------------------------------
// SA3: per-point MLP, global max over S points/batch. One wave per 16 rows.
// Weights (1.4MB f16) read as packed fragments from global (L2-resident).
// ---------------------------------------------------------------------------
__global__ void sa3_kernel(const float* __restrict__ x2, const float* __restrict__ pos2,
                           int S, int Cx,
                           const _Float16* __restrict__ W1, const float* __restrict__ b1, int C1,
                           const _Float16* __restrict__ W2, const float* __restrict__ b2, int C2,
                           const _Float16* __restrict__ W3, const float* __restrict__ b3, int C3,
                           int CinPad, int PanelC, float* __restrict__ g) {
  extern __shared__ char smem[];
  int lane = threadIdx.x & 31;
  _Float16* inP  = (_Float16*)smem;
  _Float16* outP = inP + 16 * PanelC;
  int row0 = blockIdx.x * 16;
  int b    = row0 / S;
  int Cfeat = Cx + 3;
  for (int e = lane; e < 16 * CinPad; e += 32) {
    int r = e / CinPad, c = e - r * CinPad;
    int row = row0 + r;
    float v = 0.0f;
    if (c < Cx)         v = x2[(size_t)row * Cx + c];
    else if (c < Cfeat) v = pos2[(size_t)row * 3 + (c - Cx)];
    inP[r * CinPad + c] = (_Float16)v;
  }
  __builtin_amdgcn_wave_barrier();
  mlp_layer(W1, b1, inP, outP, CinPad, C1, lane);
  mlp_layer(W2, b2, outP, inP, C1, C2, lane);
  mlp_final_max(W3, b3, inP, C2, C3, 0xFFFFu, g + (size_t)b * C3, lane);
}

// ---------------------------------------------------------------------------
// Classifier: [B=8,1024] -> 512 -> 256 -> 40. One wave; rows >= 8 zero.
// ---------------------------------------------------------------------------
__global__ void classifier_kernel(const float* __restrict__ g,
                                  const _Float16* __restrict__ W1, const float* __restrict__ b1,
                                  const _Float16* __restrict__ W2, const float* __restrict__ b2,
                                  const _Float16* __restrict__ W3, const float* __restrict__ b3,
                                  float* __restrict__ out) {
  extern __shared__ char smem[];
  int lane = threadIdx.x & 31;
  _Float16* pA = (_Float16*)smem;       // 16 x 1024
  _Float16* pB = pA + 16 * 1024;        // 16 x 512
  for (int e = lane; e < 16 * 1024; e += 32) {
    int r = e >> 10, c = e & 1023;
    pA[e] = (_Float16)((r < 8) ? g[(size_t)r * 1024 + c] : 0.0f);
  }
  __builtin_amdgcn_wave_barrier();
  mlp_layer(W1, b1, pA, pB, 1024, 512, lane);
  mlp_layer(W2, b2, pB, pA, 512, 256, lane);
  // final 256 -> 40 (weights padded to 48 cols)
  int col = lane & 15, rbase = (lane >> 4) * 8;
  for (int nt = 0; nt < 3; ++nt) {
    v8f acc = {};
    for (int kt = 0; kt < 8; ++kt) {
      v16h a  = load_a_frag(pA, 256, kt * 32, lane);
      v16h bf = load_b_packed(W3, 3, kt, nt, lane);
      acc = __builtin_amdgcn_wmma_f32_16x16x32_f16(false, a, false, bf, (short)0, acc, false, false);
    }
    int n = nt * 16 + col;
    float bv = (n < 40) ? b3[n] : 0.0f;
#pragma unroll
    for (int r = 0; r < 8; ++r) {
      int row = rbase + r;
      if (row < 8 && n < 40) out[row * 40 + n] = acc[r] + bv;
    }
  }
}

// ---------------------------------------------------------------------------
// Host orchestration
// ---------------------------------------------------------------------------
extern "C" void kernel_launch(void* const* d_in, const int* in_sizes, int n_in,
                              void* d_out, int out_size, void* d_ws, size_t ws_size,
                              hipStream_t stream) {
  (void)in_sizes; (void)n_in; (void)out_size; (void)ws_size;
  const int B = 8, P1 = 2048, S1 = 1024, S2 = 256, K = 64;

  const float* pos = (const float*)d_in[0];
  const float* Wf[12]; const float* Bf[12];
  for (int i = 0; i < 12; ++i) { Wf[i] = (const float*)d_in[1 + 2 * i]; Bf[i] = (const float*)d_in[2 + 2 * i]; }

  // layer dims: 0-2 sa1, 3-5 sa2, 6-8 sa3, 9-11 cls
  const int Cin  [12] = {3, 64, 64,   131, 128, 128,  259, 256, 512,  1024, 512, 256};
  const int Cout [12] = {64, 64, 128, 128, 128, 256,  256, 512, 1024, 512,  256, 40};
  const int CinP [12] = {32, 64, 64,  160, 128, 128,  288, 256, 512,  1024, 512, 256};
  const int CoutP[12] = {64, 64, 128, 128, 128, 256,  256, 512, 1024, 512,  256, 48};
  int Wsz[12];
  for (int i = 0; i < 12; ++i) Wsz[i] = CinP[i] * CoutP[i];

  char* ws = (char*)d_ws;
  size_t off = 0;
  auto alloc = [&](size_t bytes) -> char* {
    char* p = ws + off;
    off = (off + bytes + 255) & ~(size_t)255;
    return p;
  };
  int*   idx1 = (int*)  alloc((size_t)B * S1 * 4);
  float* pos1 = (float*)alloc((size_t)B * S1 * 3 * 4);
  int*   nb1  = (int*)  alloc((size_t)B * S1 * K * 4);
  float* dk1  = (float*)alloc((size_t)B * S1 * K * 4);
  float* x1   = (float*)alloc((size_t)B * S1 * 128 * 4);
  int*   idx2 = (int*)  alloc((size_t)B * S2 * 4);
  float* pos2 = (float*)alloc((size_t)B * S2 * 3 * 4);
  int*   nb2  = (int*)  alloc((size_t)B * S2 * K * 4);
  float* dk2  = (float*)alloc((size_t)B * S2 * K * 4);
  float* x2   = (float*)alloc((size_t)B * S2 * 256 * 4);
  float* g    = (float*)alloc((size_t)B * 1024 * 4);
  _Float16* Wh[12];
  for (int i = 0; i < 12; ++i) Wh[i] = (_Float16*)alloc((size_t)Wsz[i] * 2);

  // weight conversion f32 -> packed-fragment f16
  for (int i = 0; i < 12; ++i) {
    int nfrag = (CinP[i] >> 5) * (CoutP[i] >> 4) * 32;
    convert_w_packed<<<(nfrag + 255) / 256, 256, 0, stream>>>(Wf[i], Wh[i], Cin[i], Cout[i], CinP[i], CoutP[i]);
  }
  // init max-pool accumulators
  const float NEG = -3.4e38f;
  { int n = B * S1 * 128; fill_kernel<<<(n + 255) / 256, 256, 0, stream>>>(x1, NEG, n); }
  { int n = B * S2 * 256; fill_kernel<<<(n + 255) / 256, 256, 0, stream>>>(x2, NEG, n); }
  { int n = B * 1024;     fill_kernel<<<(n + 255) / 256, 256, 0, stream>>>(g,  NEG, n); }

  // ---- SA1 ----
  fps_kernel<<<B, 256, (P1 + 512) * sizeof(float), stream>>>(pos, P1, S1, idx1);
  gather_pos_kernel<<<(B * S1 + 255) / 256, 256, 0, stream>>>(pos, idx1, P1, S1, B * S1, pos1);
  knn_kernel<<<(B * S1) / 4, 128, (size_t)4 * P1 * sizeof(float), stream>>>(pos, P1, pos1, S1, B * S1, nb1, dk1);
  {
    size_t smem = ((size_t)(Wsz[0] + Wsz[1] + Wsz[2]) + (size_t)4 * 2 * 16 * 64) * sizeof(_Float16);
    sa_mlp_kernel<<<B * S1, 128, smem, stream>>>(
        pos, P1, nullptr, 0, idx1, S1, nb1, dk1, 0.2f * 0.2f,
        Wh[0], Bf[0], 64, Wsz[0], Wh[1], Bf[1], 64, Wsz[1], Wh[2], Bf[2], 128, Wsz[2],
        32, 64, x1);
  }

  // ---- SA2 ----
  fps_kernel<<<B, 256, (S1 + 512) * sizeof(float), stream>>>(pos1, S1, S2, idx2);
  gather_pos_kernel<<<(B * S2 + 255) / 256, 256, 0, stream>>>(pos1, idx2, S1, S2, B * S2, pos2);
  knn_kernel<<<(B * S2) / 4, 128, (size_t)4 * S1 * sizeof(float), stream>>>(pos1, S1, pos2, S2, B * S2, nb2, dk2);
  {
    size_t smem = ((size_t)(Wsz[3] + Wsz[4] + Wsz[5]) + (size_t)4 * 2 * 16 * 160) * sizeof(_Float16);
    sa_mlp_kernel<<<B * S2, 128, smem, stream>>>(
        pos1, S1, x1, 128, idx2, S2, nb2, dk2, 0.4f * 0.4f,
        Wh[3], Bf[3], 128, Wsz[3], Wh[4], Bf[4], 128, Wsz[4], Wh[5], Bf[5], 256, Wsz[5],
        160, 160, x2);
  }

  // ---- SA3 (global) ----
  sa3_kernel<<<(B * S2) / 16, 32, (size_t)2 * 16 * 512 * sizeof(_Float16), stream>>>(
      x2, pos2, S2, 256,
      Wh[6], Bf[6], 256, Wh[7], Bf[7], 512, Wh[8], Bf[8], 1024, 288, 512, g);

  // ---- Classifier ----
  classifier_kernel<<<1, 32, (size_t)(16 * 1024 + 16 * 512) * sizeof(_Float16), stream>>>(
      g, Wh[9], Bf[9], Wh[10], Bf[10], Wh[11], Bf[11], (float*)d_out);
}
